// GraphConvLayer_14370960572897
// MI455X (gfx1250) — matrix operand order repaired and checked by reading, not compiled, not run on hardware
//
#include <hip/hip_runtime.h>

// ---------- problem constants ----------
static constexpr int N_NODES  = 50000;
static constexpr int N_EDGES  = 800000;
static constexpr int NODE_DIM = 128;
static constexpr int EDGE_DIM = 64;

// ---------- vector types ----------
typedef __attribute__((ext_vector_type(16))) __bf16 v16bf;
typedef __attribute__((ext_vector_type(8)))  __bf16 v8bf;
typedef __attribute__((ext_vector_type(8)))  float  v8f;
typedef __attribute__((ext_vector_type(4)))  float  v4f;
typedef __attribute__((ext_vector_type(4)))  unsigned int u32x4;
typedef __attribute__((ext_vector_type(8)))  int    i32x8;
typedef __attribute__((ext_vector_type(4)))  int    i32x4;

// convert 8 contiguous f32 -> 8 bf16 (32B-aligned source)
__device__ __forceinline__ v8bf cvt8(const float* p) {
    v4f a = *(const v4f*)p;
    v4f b = *(const v4f*)(p + 4);
    v8bf r;
#pragma unroll
    for (int i = 0; i < 4; ++i) { r[i] = (__bf16)a[i]; r[i + 4] = (__bf16)b[i]; }
    return r;
}

__device__ __forceinline__ v16bf cat16(v8bf lo, v8bf hi) {
    v16bf r;
#pragma unroll
    for (int i = 0; i < 8; ++i) { r[i] = lo[i]; r[i + 8] = hi[i]; }
    return r;
}

__device__ __forceinline__ v8f wmma_bf16(v16bf a, v16bf b, v8f c) {
    return __builtin_amdgcn_wmma_f32_16x16x32_bf16(false, a, false, b, (short)0, c, false, false);
}

__device__ __forceinline__ float leaky(float x) { return x >= 0.f ? x : 0.2f * x; }

// ---------- TDM: 1-D async copy of `nelem` bf16 elements, global -> LDS ----------
// D# per CDNA5 ISA (2-D descriptor, 1 row).
// 6-arg builtin form (clang-23 / therock-10.0 headers): groups 2/3 + extra group zeroed.
__device__ __forceinline__ void tdm_load_1d(const void* gsrc, unsigned lds_byte_off,
                                            unsigned nelem) {
    unsigned long long ga = (unsigned long long)(uintptr_t)gsrc;
    u32x4 g0;
    g0[0] = 1u;                                            // count=1 (valid), user mode
    g0[1] = lds_byte_off;                                  // lds_addr
    g0[2] = (unsigned)(ga & 0xFFFFFFFFu);                  // global_addr[31:0]
    g0[3] = (unsigned)((ga >> 32) & 0x1FFFFFFu) | (2u << 30); // global_addr[56:32], type=2
    i32x8 g1;
    g1[0] = (int)(1u << 16);                               // data_size=1 (2 bytes)
    g1[1] = (int)((nelem & 0xFFFFu) << 16);                // tensor_dim0[15:0]
    g1[2] = (int)(((nelem >> 16) & 0xFFFFu) | (1u << 16)); // tensor_dim0[31:16], tensor_dim1=1
    g1[3] = (int)((nelem & 0xFFFFu) << 16);                // tile_dim0 = nelem
    g1[4] = 1;                                             // tile_dim1 = 1, tile_dim2 = 0
    g1[5] = (int)nelem;                                    // tensor_dim0_stride[31:0]
    g1[6] = (int)((nelem & 0xFFFFu) << 16);                // stride0 hi=0, stride1[15:0]
    g1[7] = (int)(nelem >> 16);                            // stride1[47:16]
    i32x4 z4 = {0, 0, 0, 0};
    i32x8 z8 = {0, 0, 0, 0, 0, 0, 0, 0};
    __builtin_amdgcn_tensor_load_to_lds(g0, g1, z4, z4, z8, 0);
}

// ---------- kernel 0: zero aggregation buffer (aliased onto d_out) ----------
__global__ void zero_kernel(float* __restrict__ p, int n) {
    int i = blockIdx.x * blockDim.x + threadIdx.x;
    if (i < n) p[i] = 0.f;
}

// ---------- kernel 1: pack weights -> column-major bf16 (Wt[n][k] = W[k][n]) ----------
__global__ void pack_kernel(const float* __restrict__ w1, const float* __restrict__ w2,
                            const float* __restrict__ u1, const float* __restrict__ u2,
                            __bf16* __restrict__ w1t, __bf16* __restrict__ w2t,
                            __bf16* __restrict__ u1t, __bf16* __restrict__ u2t) {
    int i = blockIdx.x * blockDim.x + threadIdx.x;
    if (i < 24576) {                       // msg_w1: (192,128) -> w1t[128][192]
        int n = i / 192, k = i % 192;
        w1t[i] = (__bf16)w1[k * 128 + n];
    } else if (i < 40960) {                // msg_w2: (128,128) -> w2t[128][128]
        int j = i - 24576; int n = j / 128, k = j % 128;
        w2t[j] = (__bf16)w2[k * 128 + n];
    } else if (i < 73728) {                // upd_w1: (256,128) -> u1t[128][256]
        int j = i - 40960; int n = j / 256, k = j % 256;
        u1t[j] = (__bf16)u1[k * 128 + n];
    } else if (i < 90112) {                // upd_w2: (128,128) -> u2t[128][128]
        int j = i - 73728; int n = j / 128, k = j % 128;
        u2t[j] = (__bf16)u2[k * 128 + n];
    }
}

// ---------- kernel 2: edge MLP + scatter-add ----------
// 8 waves / 256-thread block, one 16-edge tile per wave. 6250 blocks * 8 = 50000 tiles.
// Dynamic LDS layout (bytes):
//   [0,      49152)  w1t  (24576 bf16)
//   [49152,  81920)  w2t  (16384 bf16)
//   [81920, 147456)  h tiles: 8 waves * 16*128 bf16
//   [147456,147968)  src rows: 8 waves * 16 int
__global__ __launch_bounds__(256) void edge_kernel(
    const float* __restrict__ node_feats, const float* __restrict__ edge_feats,
    const float* __restrict__ msg_b1, const float* __restrict__ msg_b2,
    const int* __restrict__ edge_index,          // [2, E] flat: src then dst
    const __bf16* __restrict__ gw1t, const __bf16* __restrict__ gw2t,
    float* __restrict__ agg)
{
    extern __shared__ char dynsmem[];
    __bf16* lds   = (__bf16*)dynsmem;
    const unsigned lds_base = (unsigned)(uintptr_t)dynsmem;   // LDS byte offset (addr[31:0])

    const int wave = threadIdx.x >> 5;
    const int lane = threadIdx.x & 31;
    const int r    = lane & 15;
    const int half = lane >> 4;

    // ---- stage weights into LDS via Tensor Data Mover (one wave issues) ----
    if (wave == 0) {
        tdm_load_1d(gw1t, lds_base + 0,     24576);
        tdm_load_1d(gw2t, lds_base + 49152, 16384);
        __builtin_amdgcn_s_wait_tensorcnt(0);
    }
    __syncthreads();

    const __bf16* w1 = lds;                    // [128][192]
    const __bf16* w2 = lds + 24576;            // [128][128]
    __bf16* hbuf     = lds + 40960 + wave * 2048;   // 16*128 per wave
    int* lds_src     = (int*)(dynsmem + 147456) + wave * 16;

    const int tile = blockIdx.x * 8 + wave;    // [0, 50000)
    const int e    = tile * 16 + r;

    const int dst = edge_index[N_EDGES + e];
    if (half == 0) lds_src[r] = edge_index[e];

    const float* nrow = node_feats + (long long)dst * NODE_DIM;
    const float* erow = edge_feats + (long long)e * EDGE_DIM;

    // ---- A fragments: 16x192 input tile, 6 chunks of K=32 ----
    v16bf afr[6];
#pragma unroll
    for (int kc = 0; kc < 6; ++kc) {
        int k0 = kc * 32 + half * 8;
        int k1 = kc * 32 + 16 + half * 8;
        const float* p0 = (k0 < 128) ? (nrow + k0) : (erow + (k0 - 128));
        const float* p1 = (k1 < 128) ? (nrow + k1) : (erow + (k1 - 128));
        afr[kc] = cat16(cvt8(p0), cvt8(p1));
    }

    // ---- layer 1: (16x192)@(192x128) + b1, LeakyReLU -> h in LDS ----
#pragma unroll
    for (int nc = 0; nc < 8; ++nc) {
        const int n = nc * 16 + r;
        const float bv = msg_b1[n];
        v8f acc = {bv, bv, bv, bv, bv, bv, bv, bv};
        v16bf bfr[6];
#pragma unroll
        for (int kc = 0; kc < 6; ++kc)
            bfr[kc] = cat16(*(const v8bf*)(w1 + n * 192 + kc * 32 + half * 8), *(const v8bf*)(w1 + n * 192 + kc * 32 + 16 + half * 8));
#pragma unroll
        for (int kc = 0; kc < 6; ++kc)
            acc = wmma_bf16(afr[kc], bfr[kc], acc);
#pragma unroll
        for (int i = 0; i < 8; ++i)
            hbuf[(i + half * 8) * 128 + n] = (__bf16)leaky(acc[i]);
    }
    __builtin_amdgcn_wave_barrier();

    // ---- re-load h (16x128) as A fragments from LDS ----
    v16bf hfr[4];
    const __bf16* hrow = hbuf + r * 128;
#pragma unroll
    for (int kc = 0; kc < 4; ++kc) {
        v8bf lo = *(const v8bf*)(hrow + kc * 32 + half * 8);
        v8bf hi = *(const v8bf*)(hrow + kc * 32 + 16 + half * 8);
        hfr[kc] = cat16(lo, hi);
    }

    int srcRow[8];
#pragma unroll
    for (int v = 0; v < 8; ++v) srcRow[v] = lds_src[half * 8 + v];

    // ---- layer 2: (16x128)@(128x128) + b2, scatter-add to agg ----
#pragma unroll
    for (int nc = 0; nc < 8; ++nc) {
        const int n = nc * 16 + r;
        const float bv = msg_b2[n];
        v8f acc = {bv, bv, bv, bv, bv, bv, bv, bv};
        v16bf bfr[4];
#pragma unroll
        for (int kc = 0; kc < 4; ++kc)
            bfr[kc] = cat16(*(const v8bf*)(w2 + n * 128 + kc * 32 + half * 8), *(const v8bf*)(w2 + n * 128 + kc * 32 + 16 + half * 8));
#pragma unroll
        for (int kc = 0; kc < 4; ++kc)
            acc = wmma_bf16(hfr[kc], bfr[kc], acc);
#pragma unroll
        for (int v = 0; v < 8; ++v)
            unsafeAtomicAdd(&agg[(long long)srcRow[v] * 128 + n], acc[v]);
    }
}

// ---------- kernel 3: node MLP (in-place: agg buffer == out buffer) ----------
// Dynamic LDS layout (bytes):
//   [0,      65536)  u1t (32768 bf16)
//   [65536,  98304)  u2t (16384 bf16)
//   [98304, 163840)  h tiles: 8 waves * 16*128 bf16
__global__ __launch_bounds__(256) void node_kernel(
    const float* __restrict__ node_feats,
    const float* __restrict__ upd_b1, const float* __restrict__ upd_b2,
    const __bf16* __restrict__ gu1t, const __bf16* __restrict__ gu2t,
    float* data /* reads agg rows, writes out rows (same 16 rows per wave) */)
{
    extern __shared__ char dynsmem[];
    __bf16* lds   = (__bf16*)dynsmem;
    const unsigned lds_base = (unsigned)(uintptr_t)dynsmem;

    const int wave = threadIdx.x >> 5;
    const int lane = threadIdx.x & 31;
    const int r    = lane & 15;
    const int half = lane >> 4;

    if (wave == 0) {
        tdm_load_1d(gu1t, lds_base + 0,     32768);
        tdm_load_1d(gu2t, lds_base + 65536, 16384);
        __builtin_amdgcn_s_wait_tensorcnt(0);
    }
    __syncthreads();

    const int tile = blockIdx.x * 8 + wave;
    if (tile >= N_NODES / 16) return;              // wave-uniform exit after barrier

    const __bf16* u1 = lds;                        // [128][256]
    const __bf16* u2 = lds + 32768;                // [128][128]
    __bf16* hbuf     = lds + 49152 + wave * 2048;

    const int row = tile * 16 + r;
    const float* nrow = node_feats + (long long)row * NODE_DIM;
    const float* arow = data + (long long)row * NODE_DIM;

    // ---- A fragments: 16x256 concat(node, agg), 8 chunks of K=32 ----
    v16bf afr[8];
#pragma unroll
    for (int kc = 0; kc < 8; ++kc) {
        int k0 = kc * 32 + half * 8;
        int k1 = kc * 32 + 16 + half * 8;
        const float* p0 = (k0 < 128) ? (nrow + k0) : (arow + (k0 - 128));
        const float* p1 = (k1 < 128) ? (nrow + k1) : (arow + (k1 - 128));
        afr[kc] = cat16(cvt8(p0), cvt8(p1));
    }

    // ---- layer 1: (16x256)@(256x128) + b1, LeakyReLU -> LDS ----
#pragma unroll
    for (int nc = 0; nc < 8; ++nc) {
        const int n = nc * 16 + r;
        const float bv = upd_b1[n];
        v8f acc = {bv, bv, bv, bv, bv, bv, bv, bv};
        v16bf bfr[8];
#pragma unroll
        for (int kc = 0; kc < 8; ++kc)
            bfr[kc] = cat16(*(const v8bf*)(u1 + n * 256 + kc * 32 + half * 8), *(const v8bf*)(u1 + n * 256 + kc * 32 + 16 + half * 8));
#pragma unroll
        for (int kc = 0; kc < 8; ++kc)
            acc = wmma_bf16(afr[kc], bfr[kc], acc);
#pragma unroll
        for (int i = 0; i < 8; ++i)
            hbuf[(i + half * 8) * 128 + n] = (__bf16)leaky(acc[i]);
    }
    __builtin_amdgcn_wave_barrier();

    v16bf hfr[4];
    const __bf16* hrow = hbuf + r * 128;
#pragma unroll
    for (int kc = 0; kc < 4; ++kc) {
        v8bf lo = *(const v8bf*)(hrow + kc * 32 + half * 8);
        v8bf hi = *(const v8bf*)(hrow + kc * 32 + 16 + half * 8);
        hfr[kc] = cat16(lo, hi);
    }

    // ---- layer 2: (16x128)@(128x128) + b2 -> out (in place) ----
#pragma unroll
    for (int nc = 0; nc < 8; ++nc) {
        const int n = nc * 16 + r;
        const float bv = upd_b2[n];
        v8f acc = {bv, bv, bv, bv, bv, bv, bv, bv};
        v16bf bfr[4];
#pragma unroll
        for (int kc = 0; kc < 4; ++kc)
            bfr[kc] = cat16(*(const v8bf*)(u2 + n * 128 + kc * 32 + half * 8), *(const v8bf*)(u2 + n * 128 + kc * 32 + 16 + half * 8));
#pragma unroll
        for (int kc = 0; kc < 4; ++kc)
            acc = wmma_bf16(hfr[kc], bfr[kc], acc);
#pragma unroll
        for (int v = 0; v < 8; ++v)
            data[(long long)(tile * 16 + half * 8 + v) * 128 + n] = acc[v];
    }
}

// ---------- host launcher ----------
extern "C" void kernel_launch(void* const* d_in, const int* in_sizes, int n_in,
                              void* d_out, int out_size, void* d_ws, size_t ws_size,
                              hipStream_t stream) {
    const float* node_feats = (const float*)d_in[0];
    const float* edge_feats = (const float*)d_in[1];
    const float* msg_w1     = (const float*)d_in[2];
    const float* msg_b1     = (const float*)d_in[3];
    const float* msg_w2     = (const float*)d_in[4];
    const float* msg_b2     = (const float*)d_in[5];
    const float* upd_w1     = (const float*)d_in[6];
    const float* upd_b1     = (const float*)d_in[7];
    const float* upd_w2     = (const float*)d_in[8];
    const float* upd_b2     = (const float*)d_in[9];
    const int*   edge_index = (const int*)d_in[10];

    float*  out  = (float*)d_out;        // doubles as the aggregation buffer
    __bf16* wpk  = (__bf16*)d_ws;        // packed bf16 transposed weights (~176 KB)
    __bf16* w1t  = wpk;                  // 128*192
    __bf16* w2t  = wpk + 24576;          // 128*128
    __bf16* u1t  = wpk + 40960;          // 128*256
    __bf16* u2t  = wpk + 73728;          // 128*128

    const int total = N_NODES * NODE_DIM;
    zero_kernel<<<(total + 255) / 256, 256, 0, stream>>>(out, total);
    pack_kernel<<<(90112 + 255) / 256, 256, 0, stream>>>(msg_w1, msg_w2, upd_w1, upd_w2,
                                                         w1t, w2t, u1t, u2t);
    edge_kernel<<<N_EDGES / 16 / 8, 256, 147968, stream>>>(node_feats, edge_feats,
                                                           msg_b1, msg_b2, edge_index,
                                                           w1t, w2t, out);
    node_kernel<<<(N_NODES / 16 + 7) / 8, 256, 163840, stream>>>(node_feats, upd_b1, upd_b2,
                                                                 u1t, u2t, out);
}
